// VectorQuantizer_83588653514885
// MI455X (gfx1250) — matrix operand order repaired and checked by reading, not compiled, not run on hardware
//
#include <hip/hip_runtime.h>
#include <hip/hip_bf16.h>
#include <math.h>

typedef __attribute__((ext_vector_type(16))) _Float16 v16h;
typedef __attribute__((ext_vector_type(8)))  _Float16 v8h;
typedef __attribute__((ext_vector_type(8)))  float    v8f;
typedef __attribute__((ext_vector_type(4)))  float    v4f;

#define B_   1024
#define L_   64
#define V_   2048
#define DZ_  64
#define ROWS (B_ * L_)      // 65536 rows of (DZ) latent
#define NT   (V_ / 16)      // 128 N-tiles of 16 codes
#define BETA 0.25f

// ---------------- ws layout (bytes) ----------------
// [0,       262144) : codebook as f16           (2048*64*2)
// [262144,  270336) : c2[v] = ||codebook_v||^2  (2048 f32)
// [270336,  278528) : pyAcc[v] accumulators     (2048 f32)
// [278528,  278536) : scal[0]=sum y*log2(y+eps), scal[1]=sum ||val-lat||^2
// ---------------------------------------------------

// ------- d_out layout (float elements, reference tuple order) -------
// latent : [0,          4194304)
// embed  : [4194304,    8388608)
// code   : [8388608,    8454144)
// y_hard : [8454144,  142671872)
// Hy     : 142671872 ; Hyx : 142671873 ; loss : 142671874
// --------------------------------------------------------------------

__global__ __launch_bounds__(32)
void vq_prep(const float* __restrict__ cb, _Float16* __restrict__ cbh,
             float* __restrict__ c2) {
  const int v = blockIdx.x;
  const int t = threadIdx.x;                         // 32 lanes, 2 elems each
  const float2 x = *(const float2*)(cb + (size_t)v * DZ_ + 2 * t);
  cbh[(size_t)v * DZ_ + 2 * t]     = (_Float16)x.x;
  cbh[(size_t)v * DZ_ + 2 * t + 1] = (_Float16)x.y;
  float s = x.x * x.x + x.y * x.y;
  for (int o = 16; o > 0; o >>= 1) s += __shfl_down(s, o, 32);
  if (t == 0) c2[v] = s;
}

__global__ void vq_zero(float* __restrict__ pyAcc, float* __restrict__ scal) {
  const int i = blockIdx.x * blockDim.x + threadIdx.x;
  if (i < V_) pyAcc[i] = 0.0f;
  if (i < 2)  scal[i]  = 0.0f;
}

__global__ __launch_bounds__(32)
void vq_main(const float* __restrict__ q, const float* __restrict__ cb,
             const _Float16* __restrict__ cbh, const float* __restrict__ c2,
             float* __restrict__ pyAcc, float* __restrict__ scal,
             float* __restrict__ outLatent, float* __restrict__ outEmb,
             float* __restrict__ outCode, float* __restrict__ outYh) {
  __shared__ float sc[16][17];   // padded: conflict-free row scans
  __shared__ float l2s[16];
  __shared__ float Mld[16], Sld[16];
  __shared__ int   codeLd[16];

  const int lane = threadIdx.x;
  const int R0   = blockIdx.x * 16;   // 16 latent rows per workgroup
  const int hi   = lane >> 4;
  const int lo   = lane & 15;

  // ---- A tile: 16x64 latent, f16, CDNA5 16-bit A layout ----
  // lanes 0-15: row=lane, K in {kb..kb+7} U {kb+16..kb+23}, kb = hi*8
  const float* rp = q + (size_t)(R0 + lo) * DZ_;
  const int kb = hi * 8;
  v16h a0, a1;
  float l2p = 0.0f;
  #pragma unroll
  for (int j = 0; j < 8; ++j) {
    float x0 = rp[kb + j];
    float x1 = rp[kb + 16 + j];
    float x2 = rp[kb + 32 + j];
    float x3 = rp[kb + 48 + j];
    a0[j] = (_Float16)x0; a0[j + 8] = (_Float16)x1;
    a1[j] = (_Float16)x2; a1[j + 8] = (_Float16)x3;
    l2p += x0 * x0 + x1 * x1 + x2 * x2 + x3 * x3;   // exact f32 row norms
  }
  l2p += __shfl_xor(l2p, 16, 32);                   // lane pair covers full row
  if (hi == 0) l2s[lo] = l2p;
  __syncthreads();
  float l2r[8];
  #pragma unroll
  for (int i = 0; i < 8; ++i) l2r[i] = l2s[hi * 8 + i];  // C rows m = hi*8+i

  const int kbB = hi * 8;   // B layout: lane col=lo, K in {kb..kb+7} U {kb+16..kb+23}, same as A

  // ================= Sweep 1: online max / argmax / sum-exp =================
  float m = -3.0e38f, ssum = 0.0f, bestSc = -3.0e38f;
  int bestV = 0;
  for (int t = 0; t < NT; ++t) {
    const int v = t * 16 + lo;
    const _Float16* bp = cbh + (size_t)v * DZ_ + kbB;
    v8h b0l = *(const v8h*)(bp);
    v8h b0h = *(const v8h*)(bp + 16);
    v8h b1l = *(const v8h*)(bp + 32);
    v8h b1h = *(const v8h*)(bp + 48);
    v16h b0 = __builtin_shufflevector(b0l, b0h, 0,1,2,3,4,5,6,7,8,9,10,11,12,13,14,15);
    v16h b1 = __builtin_shufflevector(b1l, b1h, 0,1,2,3,4,5,6,7,8,9,10,11,12,13,14,15);
    v8f c = {};
    c = __builtin_amdgcn_wmma_f32_16x16x32_f16(false, a0, false, b0, (short)0, c, false, false);
    c = __builtin_amdgcn_wmma_f32_16x16x32_f16(false, a1, false, b1, (short)0, c, false, false);
    const float c2v = c2[v];
    #pragma unroll
    for (int i = 0; i < 8; ++i)
      sc[hi * 8 + i][lo] = c[i] - 0.5f * l2r[i] - 0.5f * c2v;  // tau = 1
    __syncthreads();
    if (hi == 0) {  // lane `lo` owns row `lo`
      float tmax = -3.0e38f; int tidx = 0;
      #pragma unroll
      for (int n = 0; n < 16; ++n) {
        float x = sc[lo][n];
        if (x > tmax) { tmax = x; tidx = t * 16 + n; }  // first-max, like argmax
      }
      float mnew  = fmaxf(m, tmax);
      float scale = __expf(m - mnew);
      float add = 0.0f;
      #pragma unroll
      for (int n = 0; n < 16; ++n) add += __expf(sc[lo][n] - mnew);
      ssum = ssum * scale + add;
      if (tmax > bestSc) { bestSc = tmax; bestV = tidx; }
      m = mnew;
    }
    __syncthreads();
  }
  if (hi == 0) {
    Mld[lo] = m; Sld[lo] = ssum; codeLd[lo] = bestV;
    outCode[R0 + lo] = (float)bestV;
  }
  __syncthreads();

  // ================= Sweep 2: y -> py atomics + Hyx total =================
  float Mr[8], Sinv[8];
  #pragma unroll
  for (int i = 0; i < 8; ++i) {
    Mr[i]   = Mld[hi * 8 + i];
    Sinv[i] = 1.0f / Sld[hi * 8 + i];
  }
  float entAcc = 0.0f;
  for (int t = 0; t < NT; ++t) {
    const int v = t * 16 + lo;
    const _Float16* bp = cbh + (size_t)v * DZ_ + kbB;
    v8h b0l = *(const v8h*)(bp);
    v8h b0h = *(const v8h*)(bp + 16);
    v8h b1l = *(const v8h*)(bp + 32);
    v8h b1h = *(const v8h*)(bp + 48);
    v16h b0 = __builtin_shufflevector(b0l, b0h, 0,1,2,3,4,5,6,7,8,9,10,11,12,13,14,15);
    v16h b1 = __builtin_shufflevector(b1l, b1h, 0,1,2,3,4,5,6,7,8,9,10,11,12,13,14,15);
    v8f c = {};
    c = __builtin_amdgcn_wmma_f32_16x16x32_f16(false, a0, false, b0, (short)0, c, false, false);
    c = __builtin_amdgcn_wmma_f32_16x16x32_f16(false, a1, false, b1, (short)0, c, false, false);
    const float c2v = c2[v];
    float colSum = 0.0f;
    #pragma unroll
    for (int i = 0; i < 8; ++i) {
      float s = c[i] - 0.5f * l2r[i] - 0.5f * c2v;       // identical to sweep 1
      float y = __expf(s - Mr[i]) * Sinv[i];
      colSum += y;
      entAcc += y * __log2f(y + 1e-6f);
    }
    float tot = colSum + __shfl_xor(colSum, 16, 32);     // both halves of col v
    if (hi == 0) atomicAdd(pyAcc + v, tot);
  }
  for (int o = 16; o > 0; o >>= 1) entAcc += __shfl_down(entAcc, o, 32);
  if (lane == 0) atomicAdd(scal + 0, entAcc);

  // ================= y_hard: one-hot rows, NT vector stores =================
  for (int r = 0; r < 16; ++r) {
    const int cidx = codeLd[r];
    v4f* yr4 = (v4f*)(outYh + (size_t)(R0 + r) * V_);
    #pragma unroll
    for (int kk = 0; kk < 16; ++kk) {            // 32 lanes * 16 = 512 v4f
      const int e0 = (lane + 32 * kk) * 4;
      v4f val;
      val.x = (cidx == e0 + 0) ? 1.0f : 0.0f;
      val.y = (cidx == e0 + 1) ? 1.0f : 0.0f;
      val.z = (cidx == e0 + 2) ? 1.0f : 0.0f;
      val.w = (cidx == e0 + 3) ? 1.0f : 0.0f;
      __builtin_nontemporal_store(val, yr4 + lane + 32 * kk);
    }
  }

  // ====== latent copy, embedding gather (f32 codebook), loss total ======
  float lossAcc = 0.0f;
  for (int r = 0; r < 16; ++r) {
    const int R = R0 + r;
    const int cidx = codeLd[r];
    const float2 lv = *(const float2*)(q  + (size_t)R    * DZ_ + 2 * lane);
    const float2 cv = *(const float2*)(cb + (size_t)cidx * DZ_ + 2 * lane);
    *(float2*)(outLatent + (size_t)R * DZ_ + 2 * lane) = lv;  // STE fwd == values for emb
    *(float2*)(outEmb    + (size_t)R * DZ_ + 2 * lane) = cv;
    float dx = cv.x - lv.x, dy = cv.y - lv.y;
    lossAcc += dx * dx + dy * dy;
  }
  for (int o = 16; o > 0; o >>= 1) lossAcc += __shfl_down(lossAcc, o, 32);
  if (lane == 0) atomicAdd(scal + 1, lossAcc);
}

__global__ void vq_final(const float* __restrict__ pyAcc, const float* __restrict__ scal,
                         float* __restrict__ outHy, float* __restrict__ outHyx,
                         float* __restrict__ outLoss) {
  __shared__ float red[256];
  const int t = threadIdx.x;
  float acc = 0.0f;
  for (int v = t; v < V_; v += 256) {
    float p = pyAcc[v] * (1.0f / (float)ROWS);
    acc += p * __log2f(p + 1e-6f);
  }
  red[t] = acc;
  __syncthreads();
  for (int s = 128; s > 0; s >>= 1) {
    if (t < s) red[t] += red[t + s];
    __syncthreads();
  }
  if (t == 0) {
    *outHy   = -red[0];
    *outHyx  = -scal[0] * (1.0f / (float)ROWS);
    *outLoss = (1.0f + BETA) * scal[1] * (1.0f / (float)ROWS);  // fwd: both terms equal
  }
}

extern "C" void kernel_launch(void* const* d_in, const int* in_sizes, int n_in,
                              void* d_out, int out_size, void* d_ws, size_t ws_size,
                              hipStream_t stream) {
  const float* q  = (const float*)d_in[0];   // (B, L*DZ) f32
  const float* cb = (const float*)d_in[1];   // (V, DZ)   f32
  float* out = (float*)d_out;
  char*  ws  = (char*)d_ws;

  _Float16* cbh   = (_Float16*)ws;
  float*    c2    = (float*)(ws + 262144);
  float*    pyAcc = (float*)(ws + 270336);
  float*    scal  = (float*)(ws + 278528);

  float* outLatent = out;
  float* outEmb    = out + 4194304;
  float* outCode   = out + 8388608;
  float* outYh     = out + 8454144;
  float* outHy     = out + 142671872;
  float* outHyx    = out + 142671873;
  float* outLoss   = out + 142671874;

  vq_prep<<<V_, 32, 0, stream>>>(cb, cbh, c2);
  vq_zero<<<(V_ + 255) / 256, 256, 0, stream>>>(pyAcc, scal);
  vq_main<<<ROWS / 16, 32, 0, stream>>>(q, cb, cbh, c2, pyAcc, scal,
                                        outLatent, outEmb, outCode, outYh);
  vq_final<<<1, 256, 0, stream>>>(pyAcc, scal, outHy, outHyx, outLoss);
}
